// MultiHeadAttention_90005334655017
// MI455X (gfx1250) — matrix operand order repaired and checked by reading, not compiled, not run on hardware
//
#include <hip/hip_runtime.h>
#include <hip/hip_bf16.h>

// ---------------------------------------------------------------------------
// CDNA5 (gfx1250) fused multi-head attention, bf16 WMMA path.
//   B=8, N=1024, C=768, H=8, hd=96.
// Pipeline: cvt(x,Wqkv,Wproj)->bf16 ; GEMM(x,Wqkv)+b -> qkv(bf16) ;
//           flash-attention (online softmax, WMMA QK^T and PV) -> ao(bf16) ;
//           GEMM(ao,Wproj)+b -> out(f32).
// Cross-lane softmax reductions use DPP16 row_ror (VALU-only) instead of
// ds_bpermute; key block = 64 to amortize softmax over 24 WMMAs/iteration.
// ---------------------------------------------------------------------------

typedef __attribute__((ext_vector_type(16))) __bf16 v16bf;
typedef __attribute__((ext_vector_type(8)))  __bf16 v8bf;
typedef __attribute__((ext_vector_type(8)))  float  v8f;

static __device__ inline v16bf cat8(v8bf lo, v8bf hi) {
    v16bf r;
#pragma unroll
    for (int e = 0; e < 8; ++e) { r[e] = lo[e]; r[e + 8] = hi[e]; }
    return r;
}

// ---- DPP16 row_ror rotate within 16-lane rows (wave32: two rows) -----------
template <int CTRL>
static __device__ inline float dpp_ror(float x) {
    return __builtin_bit_cast(float,
        __builtin_amdgcn_mov_dpp(__builtin_bit_cast(int, x), CTRL, 0xf, 0xf, true));
}
// log-rotate reductions: every lane of a 16-lane row ends with the full result
static __device__ inline float redmax16(float v) {
    v = fmaxf(v, dpp_ror<0x121>(v));   // row_ror:1
    v = fmaxf(v, dpp_ror<0x122>(v));   // row_ror:2
    v = fmaxf(v, dpp_ror<0x124>(v));   // row_ror:4
    v = fmaxf(v, dpp_ror<0x128>(v));   // row_ror:8
    return v;
}
static __device__ inline float redsum16(float v) {
    v += dpp_ror<0x121>(v);
    v += dpp_ror<0x122>(v);
    v += dpp_ror<0x124>(v);
    v += dpp_ror<0x128>(v);
    return v;
}

// ---------------------------------------------------------------------------
// fp32 -> bf16 conversion (vectorized x4)
// ---------------------------------------------------------------------------
__global__ __launch_bounds__(256) void cvt_f32_bf16(const float* __restrict__ in,
                                                    __bf16* __restrict__ out, int n4) {
    int i = blockIdx.x * 256 + threadIdx.x;
    if (i < n4) {
        float4 v = ((const float4*)in)[i];
        out[4 * i + 0] = (__bf16)v.x;
        out[4 * i + 1] = (__bf16)v.y;
        out[4 * i + 2] = (__bf16)v.z;
        out[4 * i + 3] = (__bf16)v.w;
    }
}

// ---------------------------------------------------------------------------
// bf16 GEMM: C[M,Nn] = A[M,K] * Bw[K,Nn] + bias[Nn]
// Block = 256 threads (8 waves), tile BM=128 x BN=128, BK=32.
// Wave w -> 32x64 micro-tile: (w>>1)*32 rows, (w&1)*64 cols (2x4 WMMA frags).
// ---------------------------------------------------------------------------
constexpr int BM = 128, BN = 128, BK = 32;

template <bool OUT_BF16>
__global__ __launch_bounds__(256) void gemm_bf16(const __bf16* __restrict__ A,
                                                 const __bf16* __restrict__ Bw,
                                                 const float* __restrict__ bias,
                                                 void* __restrict__ Cout,
                                                 int M, int Nn, int K) {
    __shared__ __align__(16) __bf16 lsA[BM * BK];
    __shared__ __align__(16) __bf16 lsB[BK * BN];

    const int tid  = threadIdx.x;
    const int lane = tid & 31;
    const int wave = tid >> 5;
    const int m0 = blockIdx.y * BM;
    const int n0 = blockIdx.x * BN;
    const int wm = (wave >> 1) * 32;
    const int wn = (wave & 1) * 64;
    const int r  = lane & 15;       // A-frag row
    const int kb = lane >> 4;       // A-frag K-half selector

    v8f acc[2][4] = {};

    for (int k0 = 0; k0 < K; k0 += BK) {
        __syncthreads();
        // ---- cooperative LDS fill: A tile 128x32 (two passes of 16B/thread)
        {
            int lin = tid * 8;
#pragma unroll
            for (int p = 0; p < 2; ++p) {
                int row = lin >> 5, col = lin & 31;
                *(v8bf*)&lsA[row * BK + col] =
                    *(const v8bf*)&A[(long)(m0 + row) * K + k0 + col];
                lin += 256 * 8;
            }
        }
        // ---- B tile 32x128 (two passes of 16B/thread)
        {
            int lin = tid * 8;
#pragma unroll
            for (int p = 0; p < 2; ++p) {
                int row = lin >> 7, col = lin & 127;
                *(v8bf*)&lsB[row * BN + col] =
                    *(const v8bf*)&Bw[(long)(k0 + row) * Nn + n0 + col];
                lin += 256 * 8;
            }
        }
        // ---- L2 prefetch of the next K-slab (global_prefetch_b8)
        if (k0 + BK < K) {
            int lin = tid * 8;
            __builtin_prefetch(&A[(long)(m0 + (lin >> 5)) * K + k0 + BK + (lin & 31)], 0, 1);
            __builtin_prefetch(&Bw[(long)(k0 + BK + (lin >> 7)) * Nn + n0 + (lin & 127)], 0, 1);
        }
        __syncthreads();

        // ---- fragments (ISA layouts): A lane=M, K split {kb*8..+8, 16+kb*8..+8}
        v16bf afrag[2];
#pragma unroll
        for (int i = 0; i < 2; ++i) {
            const __bf16* ap = &lsA[(wm + i * 16 + r) * BK + kb * 8];
            afrag[i] = cat8(*(const v8bf*)ap, *(const v8bf*)(ap + 16));
        }
#pragma unroll
        for (int j = 0; j < 4; ++j) {
            // B: lane = N column, K split {kb*8..+8, 16+kb*8..+8}
            v16bf bfrag;
#pragma unroll
            for (int e = 0; e < 16; ++e)
                bfrag[e] = lsB[((e & 8) * 2 + kb * 8 + (e & 7)) * BN + wn + j * 16 + r];
#pragma unroll
            for (int i = 0; i < 2; ++i)
                acc[i][j] = __builtin_amdgcn_wmma_f32_16x16x32_bf16(
                    false, afrag[i], false, bfrag, (short)0, acc[i][j],
                    false, false);
        }
    }

    // ---- store: C/D layout lane=(N, half-wave M+8), VGPR e = M row
    const int mBase = m0 + wm + ((lane >> 4) << 3);
    const int nBase = n0 + wn + (lane & 15);
#pragma unroll
    for (int i = 0; i < 2; ++i)
#pragma unroll
        for (int j = 0; j < 4; ++j) {
            int col = nBase + j * 16;
            float bv = bias[col];
#pragma unroll
            for (int e = 0; e < 8; ++e) {
                int row = mBase + i * 16 + e;
                float v = acc[i][j][e] + bv;
                if constexpr (OUT_BF16)
                    ((__bf16*)Cout)[(long)row * Nn + col] = (__bf16)v;
                else
                    ((float*)Cout)[(long)row * Nn + col] = v;
            }
        }
}

// ---------------------------------------------------------------------------
// Flash attention: grid (N/128, H, B), 256 threads (8 waves).
// Wave owns 16 query rows; key blocks of 64 (24 WMMAs per block).
// qkv layout: [B*N, 2304] with q at +0, k at +768, v at +1536, head h at +h*96.
// out: bf16 [B*N, 768] (head transpose fused into store).
// ---------------------------------------------------------------------------
constexpr int HD = 96, BKV = 64;

__global__ __launch_bounds__(256) void flash_attn(const __bf16* __restrict__ qkv,
                                                  __bf16* __restrict__ out,
                                                  int Bb, int N) {
    __shared__ __align__(16) __bf16 lsKt[HD * BKV];       // [d][key] transposed
    __shared__ __align__(16) __bf16 lsV[BKV * HD];        // [key][d]
    __shared__ __align__(16) __bf16 lsP[8][16 * BKV];     // per-wave P transpose

    const int tid  = threadIdx.x;
    const int lane = tid & 31;
    const int wave = tid >> 5;
    const int b = blockIdx.z, h = blockIdx.y;
    const int q0 = blockIdx.x * 128 + wave * 16;
    const int r  = lane & 15;
    const int kb = lane >> 4;
    const long RS = 2304;  // qkv row stride (elements)

    // log2-domain softmax: fold (hd^-0.5 * log2 e) into Q once
    const float sscale = 0.10206207261596575f * 1.4426950408889634f;

    // ---- Q fragments: 3 chunks of K=32 along hd, loaded straight from global
    const __bf16* qbase = qkv + (long)(b * N) * RS + h * 96;
    v16bf qf[3];
#pragma unroll
    for (int c = 0; c < 3; ++c) {
        const __bf16* p = qbase + (long)(q0 + r) * RS + c * 32 + kb * 8;
        v16bf q = cat8(*(const v8bf*)p, *(const v8bf*)(p + 16));
#pragma unroll
        for (int e = 0; e < 16; ++e) q[e] = (__bf16)((float)q[e] * sscale);
        qf[c] = q;
    }

    v8f acc[6] = {};
    float mrow[8], lrow[8];
#pragma unroll
    for (int e = 0; e < 8; ++e) { mrow[e] = -1.0e30f; lrow[e] = 0.0f; }

    for (int n0 = 0; n0 < N; n0 += BKV) {
        __syncthreads();
        // ---- cooperative K (transposed) + V load into LDS
        const __bf16* kbase = qkv + (long)(b * N + n0) * RS + 768 + h * 96;
        const __bf16* vbase = kbase + 768;
#pragma unroll
        for (int p = 0; p < 3; ++p) {               // 64*96 = 6144 = 3*2048
            int idx = tid * 8 + p * 2048;
            int key = idx / HD, d = idx - key * HD; // 96 % 8 == 0: no row cross
            v8bf kv = *(const v8bf*)&kbase[(long)key * RS + d];
#pragma unroll
            for (int e = 0; e < 8; ++e) lsKt[(d + e) * BKV + key] = kv[e];
            *(v8bf*)&lsV[idx] = *(const v8bf*)&vbase[(long)key * RS + d];
        }
        // ---- L2 prefetch of next key block
        if (n0 + BKV < N) {
            int idx = tid * 8;
            int key = idx / HD, d = idx - key * HD;
            __builtin_prefetch(&kbase[(long)(key + BKV) * RS + d], 0, 1);
            __builtin_prefetch(&vbase[(long)(key + BKV) * RS + d], 0, 1);
        }
        __syncthreads();

        // ---- S = Q * K^T  (four 16-key C-fragments, 3 K-chunks each)
        v8f s[4] = {};
#pragma unroll
        for (int c = 0; c < 3; ++c)
#pragma unroll
            for (int j = 0; j < 4; ++j) {
                v16bf bf;
#pragma unroll
                for (int e = 0; e < 16; ++e)
                    bf[e] = lsKt[(c * 32 + (e & 8) * 2 + kb * 8 + (e & 7)) * BKV + j * 16 + r];
                s[j] = __builtin_amdgcn_wmma_f32_16x16x32_bf16(
                    false, qf[c], false, bf, (short)0, s[j], false, false);
            }

        // ---- online softmax update (DPP row_ror reductions, VALU only)
#pragma unroll
        for (int e = 0; e < 8; ++e) {
            float rm = fmaxf(fmaxf(s[0][e], s[1][e]), fmaxf(s[2][e], s[3][e]));
            rm = redmax16(rm);
            float mn    = fmaxf(mrow[e], rm);
            float alpha = exp2f(mrow[e] - mn);
            mrow[e] = mn;
            float p0 = exp2f(s[0][e] - mn);
            float p1 = exp2f(s[1][e] - mn);
            float p2 = exp2f(s[2][e] - mn);
            float p3 = exp2f(s[3][e] - mn);
            float ps = redsum16((p0 + p1) + (p2 + p3));
            lrow[e] = lrow[e] * alpha + ps;
#pragma unroll
            for (int c = 0; c < 6; ++c) acc[c][e] *= alpha;
            // P (C-layout) -> per-wave LDS scratch, row-major [16 x 64]
            int row = e + ((lane >> 4) << 3);
            __bf16* pr = &lsP[wave][row * BKV + (lane & 15)];
            pr[0]  = (__bf16)p0;
            pr[16] = (__bf16)p1;
            pr[32] = (__bf16)p2;
            pr[48] = (__bf16)p3;
        }

        // ---- O += P * V : P re-read in A-layout (2 frags), V B-frags gathered (lane = d, K split as A)
        const __bf16* pb = &lsP[wave][r * BKV + kb * 8];
        v16bf pf0 = cat8(*(const v8bf*)pb,        *(const v8bf*)(pb + 16));
        v16bf pf1 = cat8(*(const v8bf*)(pb + 32), *(const v8bf*)(pb + 48));
#pragma unroll
        for (int c = 0; c < 6; ++c) {
            v16bf vf0, vf1;
#pragma unroll
            for (int e = 0; e < 16; ++e) {
                int kk = (e & 8) * 2 + kb * 8 + (e & 7);
                vf0[e] = lsV[kk * HD + c * 16 + r];
                vf1[e] = lsV[(32 + kk) * HD + c * 16 + r];
            }
            acc[c] = __builtin_amdgcn_wmma_f32_16x16x32_bf16(
                false, pf0, false, vf0, (short)0, acc[c], false, false);

            acc[c] = __builtin_amdgcn_wmma_f32_16x16x32_bf16(
                false, pf1, false, vf1, (short)0, acc[c], false, false);
        }
    }

    // ---- normalize and store bf16 in [B*N, 768] with head-transpose fused
#pragma unroll
    for (int e = 0; e < 8; ++e) {
        float inv = 1.0f / lrow[e];
        int row = q0 + e + ((lane >> 4) << 3);
#pragma unroll
        for (int c = 0; c < 6; ++c) {
            int col = h * 96 + c * 16 + (lane & 15);
            out[(long)(b * N + row) * 768 + col] = (__bf16)(acc[c][e] * inv);
        }
    }
}

// ---------------------------------------------------------------------------
// Launch
// ---------------------------------------------------------------------------
extern "C" void kernel_launch(void* const* d_in, const int* in_sizes, int n_in,
                              void* d_out, int out_size, void* d_ws, size_t ws_size,
                              hipStream_t stream) {
    (void)in_sizes; (void)n_in; (void)out_size; (void)ws_size;
    const int Bb = 8, N = 1024, C = 768;
    const int M = Bb * N;            // 8192
    const int C3 = 3 * C;            // 2304

    const float* x      = (const float*)d_in[0];
    const float* qkv_w  = (const float*)d_in[1];
    const float* qkv_b  = (const float*)d_in[2];
    const float* proj_w = (const float*)d_in[3];
    const float* proj_b = (const float*)d_in[4];
    float* out = (float*)d_out;

    char* ws = (char*)d_ws;
    __bf16* xb    = (__bf16*)(ws);                 // 8192*768   bf16
    __bf16* wqkv  = (__bf16*)(ws + 12582912);      // 768*2304   bf16
    __bf16* wproj = (__bf16*)(ws + 16121856);      // 768*768    bf16
    __bf16* qkvb  = (__bf16*)(ws + 17301504);      // 8192*2304  bf16
    __bf16* ao    = (__bf16*)(ws + 55050240);      // 8192*768   bf16

    // 1) fp32 -> bf16
    {
        int n4 = (M * C) / 4;
        cvt_f32_bf16<<<(n4 + 255) / 256, 256, 0, stream>>>(x, xb, n4);
        n4 = (C * C3) / 4;
        cvt_f32_bf16<<<(n4 + 255) / 256, 256, 0, stream>>>(qkv_w, wqkv, n4);
        n4 = (C * C) / 4;
        cvt_f32_bf16<<<(n4 + 255) / 256, 256, 0, stream>>>(proj_w, wproj, n4);
    }

    // 2) QKV GEMM: [8192,768] x [768,2304] + b -> bf16
    {
        dim3 grid(C3 / BN, M / BM);
        gemm_bf16<true><<<grid, 256, 0, stream>>>(xb, wqkv, qkv_b, (void*)qkvb,
                                                  M, C3, C);
    }

    // 3) Flash attention -> ao bf16 [8192,768]
    {
        dim3 grid(N / 128, 8 /*H*/, Bb);
        flash_attn<<<grid, 256, 0, stream>>>(qkvb, ao, Bb, N);
    }

    // 4) Projection GEMM: [8192,768] x [768,768] + b -> f32 out
    {
        dim3 grid(C / BN, M / BM);
        gemm_bf16<false><<<grid, 256, 0, stream>>>(ao, wproj, proj_b, (void*)out,
                                                   M, C, C);
    }
}
